// MPNN2_17257178596040
// MI455X (gfx1250) — matrix operand order repaired and checked by reading, not compiled, not run on hardware
//
#include <hip/hip_runtime.h>

#define NB 4
#define NN 4096
#define DF 128

typedef __bf16 bf16_t;
typedef __attribute__((ext_vector_type(16))) __bf16 v16bf;
typedef __attribute__((ext_vector_type(8)))  __bf16 v8bf;
typedef __attribute__((ext_vector_type(8)))  float  v8f;
typedef __attribute__((ext_vector_type(4)))  float  v4f;

static constexpr int LDT = 40;   // LDS row stride in elements for 32-wide K tiles (80B, 16B-aligned runs)

union FragU { v16bf v; v8bf h8[2]; };

__device__ __forceinline__ v8f wmma_bf(v16bf a, v16bf b, v8f c) {
  // D = A(16x32 bf16) x B(32x16 bf16) + C(16x16 f32)
  return __builtin_amdgcn_wmma_f32_16x16x32_bf16(false, a, false, b, (short)0, c, false, false);
}

// A fragment: lane m = lane&15, h = lane>>4; elems 0-7 = row[8h..8h+7], 8-15 = row[16+8h..+7]
__device__ __forceinline__ v16bf frag_A(const bf16_t* row, int h) {
  FragU f;
  f.h8[0] = *(const v8bf*)(row + 8 * h);
  f.h8[1] = *(const v8bf*)(row + 16 + 8 * h);
  return f.v;
}

// B fragment: lane n = lane&15, h = lane>>4; elems 0-7 = row[8h..8h+7], 8-15 = row[16+8h..+7]
__device__ __forceinline__ v16bf frag_B(const bf16_t* row, int h) {
  FragU f;
  f.h8[0] = *(const v8bf*)(row + 8 * h);
  f.h8[1] = *(const v8bf*)(row + 16 + 8 * h);
  return f.v;
}

__device__ __forceinline__ void split2(float v, bf16_t* hi, bf16_t* lo) {
  bf16_t h = (bf16_t)v;
  *hi = h;
  *lo = (bf16_t)(v - (float)h);
}

// -------- Kernel 1: xs = x@Ws, xr = x@Wr  (one GEMM: x @ [Ws|Wr], K=128) --------
__global__ void __launch_bounds__(128) mpnn_k1_proj(
    const float* __restrict__ x, const float* __restrict__ Wmsg,
    bf16_t* __restrict__ xs_hi, bf16_t* __restrict__ xs_lo, float* __restrict__ xr)
{
  __shared__ alignas(16) bf16_t sAh[64 * LDT];
  __shared__ alignas(16) bf16_t sAl[64 * LDT];
  __shared__ alignas(16) bf16_t sBh[256 * LDT];
  __shared__ alignas(16) bf16_t sBl[256 * LDT];

  const int t = threadIdx.x;
  const int wave = t >> 5;
  const int m = t & 15;
  const int h = (t >> 4) & 1;
  const int b = blockIdx.x >> 6;
  const int r0 = (blockIdx.x & 63) * 64;

  v8f acc[16] = {};

  for (int k0 = 0; k0 < DF; k0 += 32) {
    __syncthreads();
    // stage A: x tile 64x32 -> hi/lo
    for (int p = 0; p < 4; ++p) {
      const int idx = p * 128 + t;
      const int row = idx >> 3;
      const int kb = (idx & 7) * 4;
      const float4 v = *(const float4*)&x[((size_t)b * NN + r0 + row) * DF + k0 + kb];
      split2(v.x, &sAh[row * LDT + kb + 0], &sAl[row * LDT + kb + 0]);
      split2(v.y, &sAh[row * LDT + kb + 1], &sAl[row * LDT + kb + 1]);
      split2(v.z, &sAh[row * LDT + kb + 2], &sAl[row * LDT + kb + 2]);
      split2(v.w, &sAh[row * LDT + kb + 3], &sAl[row * LDT + kb + 3]);
    }
    // stage B: [Ws|Wr] tile 32x256, transposed into [n][k]
    for (int p = 0; p < 16; ++p) {
      const int idx = p * 128 + t;
      const int n4 = (idx & 63) * 4;
      const int k = idx >> 6;
      const float* src = (n4 < DF) ? &Wmsg[(size_t)(k0 + k) * DF + n4]
                                   : &Wmsg[(size_t)(DF + k0 + k) * DF + (n4 - DF)];
      const float4 v = *(const float4*)src;
      split2(v.x, &sBh[(n4 + 0) * LDT + k], &sBl[(n4 + 0) * LDT + k]);
      split2(v.y, &sBh[(n4 + 1) * LDT + k], &sBl[(n4 + 1) * LDT + k]);
      split2(v.z, &sBh[(n4 + 2) * LDT + k], &sBl[(n4 + 2) * LDT + k]);
      split2(v.w, &sBh[(n4 + 3) * LDT + k], &sBl[(n4 + 3) * LDT + k]);
    }
    __syncthreads();
    const v16bf ah = frag_A(&sAh[(wave * 16 + m) * LDT], h);
    const v16bf al = frag_A(&sAl[(wave * 16 + m) * LDT], h);
    #pragma unroll
    for (int ut = 0; ut < 16; ++ut) {
      const v16bf bh = frag_B(&sBh[(ut * 16 + m) * LDT], h);
      const v16bf bl = frag_B(&sBl[(ut * 16 + m) * LDT], h);
      acc[ut] = wmma_bf(ah, bh, acc[ut]);
      acc[ut] = wmma_bf(ah, bl, acc[ut]);
      acc[ut] = wmma_bf(al, bh, acc[ut]);
    }
  }
  #pragma unroll
  for (int ut = 0; ut < 16; ++ut) {
    const int u = (ut & 7) * 16 + m;
    #pragma unroll
    for (int e = 0; e < 8; ++e) {
      const int rl = wave * 16 + e + 8 * h;
      const size_t off = ((size_t)b * NN + r0 + rl) * DF + u;
      const float val = acc[ut][e];
      if (ut < 8) {
        split2(val, &xs_hi[off], &xs_lo[off]);   // xs stored pre-split for kernel 2
      } else {
        xr[off] = val;
      }
    }
  }
}

// -------- Kernel 2: snd_sum = adj^T @ xs (K=4096), software-pipelined, fused epilogue --------
__global__ void __launch_bounds__(128) mpnn_k2_gather(
    const float* __restrict__ adj,
    const bf16_t* __restrict__ xs_hi, const bf16_t* __restrict__ xs_lo,
    const float* __restrict__ xr, float* __restrict__ msg)
{
  __shared__ alignas(16) bf16_t sA [64 * LDT];
  __shared__ alignas(16) bf16_t sBh[128 * LDT];
  __shared__ alignas(16) bf16_t sBl[128 * LDT];
  __shared__ float sDegP[64 * 8];
  __shared__ float sDegF[64];

  const int t = threadIdx.x;
  const int wave = t >> 5;
  const int m = t & 15;
  const int h = (t >> 4) & 1;
  const int b = blockIdx.x >> 6;
  const int r0 = (blockIdx.x & 63) * 64;
  const int rb = (t & 15) * 4;   // adj staging: 4 consecutive receiver columns
  const int slot = t >> 4;       // adj staging: sender-row slot 0..7
  const int sxs  = t >> 4;       // xs staging: base sender row (reuse layout idx>>4 == p*8 + t>>4)
  const int ub   = (t & 15) * 8; // xs staging: 8-wide unit block

  v8f acc[8] = {};
  float dx = 0.f, dy = 0.f, dz = 0.f, dw = 0.f;

  const size_t adjB = (size_t)b * NN * NN;
  const size_t xsB  = (size_t)b * NN * DF;

  // registers for the software pipeline (next chunk's global data)
  v4f  aReg[4];
  v8bf hReg[4];
  v8bf lReg[4];

  auto load_globals = [&](int s0) {
    #pragma unroll
    for (int i = 0; i < 4; ++i) {
      const int s = slot + 8 * i;
      // adj is streamed exactly once: non-temporal so it doesn't evict the reused xs/x/W in L2
      aReg[i] = __builtin_nontemporal_load(
          (const v4f*)&adj[adjB + (size_t)(s0 + s) * NN + r0 + rb]);
    }
    #pragma unroll
    for (int p = 0; p < 4; ++p) {
      const int s = p * 8 + sxs;
      hReg[p] = *(const v8bf*)&xs_hi[xsB + (size_t)(s0 + s) * DF + ub];
      lReg[p] = *(const v8bf*)&xs_lo[xsB + (size_t)(s0 + s) * DF + ub];
    }
  };

  load_globals(0);

  for (int s0 = 0; s0 < NN; s0 += 32) {
    __syncthreads();   // previous chunk's LDS reads complete
    // store staged registers to LDS; degree accumulated for free on the adj stream
    #pragma unroll
    for (int i = 0; i < 4; ++i) {
      const int s = slot + 8 * i;
      const v4f v = aReg[i];
      dx += v[0]; dy += v[1]; dz += v[2]; dw += v[3];
      sA[(rb + 0) * LDT + s] = (bf16_t)v[0];
      sA[(rb + 1) * LDT + s] = (bf16_t)v[1];
      sA[(rb + 2) * LDT + s] = (bf16_t)v[2];
      sA[(rb + 3) * LDT + s] = (bf16_t)v[3];
    }
    #pragma unroll
    for (int p = 0; p < 4; ++p) {
      const int s = p * 8 + sxs;
      #pragma unroll
      for (int j = 0; j < 8; ++j) {
        sBh[(ub + j) * LDT + s] = hReg[p][j];
        sBl[(ub + j) * LDT + s] = lReg[p][j];
      }
    }
    __syncthreads();   // staging visible
    // issue next chunk's global loads now; they overlap the WMMA block below
    if (s0 + 32 < NN) load_globals(s0 + 32);

    const v16bf a = frag_A(&sA[(wave * 16 + m) * LDT], h);
    #pragma unroll
    for (int ut = 0; ut < 8; ++ut) {
      const v16bf bh = frag_B(&sBh[(ut * 16 + m) * LDT], h);
      const v16bf bl = frag_B(&sBl[(ut * 16 + m) * LDT], h);
      acc[ut] = wmma_bf(a, bh, acc[ut]);   // exact-adj x hi part
      acc[ut] = wmma_bf(a, bl, acc[ut]);   // exact-adj x lo part -> ~fp32 sum
    }
  }
  // reduce degree partials: deg[r] = sum over 8 slots
  sDegP[(rb + 0) * 8 + slot] = dx;
  sDegP[(rb + 1) * 8 + slot] = dy;
  sDegP[(rb + 2) * 8 + slot] = dz;
  sDegP[(rb + 3) * 8 + slot] = dw;
  __syncthreads();
  if (t < 64) {
    float d = 0.f;
    #pragma unroll
    for (int j = 0; j < 8; ++j) d += sDegP[t * 8 + j];
    sDegF[t] = d;
  }
  __syncthreads();
  // fused epilogue: messages = deg>0 ? snd/max(deg,1) + xr : 0
  #pragma unroll
  for (int ut = 0; ut < 8; ++ut) {
    const int u = ut * 16 + m;
    #pragma unroll
    for (int e = 0; e < 8; ++e) {
      const int rl = wave * 16 + e + 8 * h;
      const float deg = sDegF[rl];
      const size_t off = ((size_t)b * NN + r0 + rl) * DF + u;
      float val = 0.f;
      if (deg > 0.f) val = acc[ut][e] / fmaxf(deg, 1.f) + xr[off];
      msg[off] = val;
    }
  }
}

// -------- Kernel 3: out = [x | messages] @ W_upd  (K=256) --------
__global__ void __launch_bounds__(128) mpnn_k3_update(
    const float* __restrict__ x, const float* __restrict__ msg,
    const float* __restrict__ Wupd, float* __restrict__ out)
{
  __shared__ alignas(16) bf16_t sAh[64 * LDT];
  __shared__ alignas(16) bf16_t sAl[64 * LDT];
  __shared__ alignas(16) bf16_t sBh[128 * LDT];
  __shared__ alignas(16) bf16_t sBl[128 * LDT];

  const int t = threadIdx.x;
  const int wave = t >> 5;
  const int m = t & 15;
  const int h = (t >> 4) & 1;
  const int b = blockIdx.x >> 6;
  const int r0 = (blockIdx.x & 63) * 64;

  v8f acc[8] = {};

  for (int k0 = 0; k0 < 2 * DF; k0 += 32) {
    __syncthreads();
    // stage A: concat(x, msg) tile 64x32 -> hi/lo
    for (int p = 0; p < 4; ++p) {
      const int idx = p * 128 + t;
      const int row = idx >> 3;
      const int kb = (idx & 7) * 4;
      const int kk = k0 + kb;
      const size_t roff = ((size_t)b * NN + r0 + row) * DF;
      const float4 v = (kk < DF) ? *(const float4*)&x[roff + kk]
                                 : *(const float4*)&msg[roff + kk - DF];
      split2(v.x, &sAh[row * LDT + kb + 0], &sAl[row * LDT + kb + 0]);
      split2(v.y, &sAh[row * LDT + kb + 1], &sAl[row * LDT + kb + 1]);
      split2(v.z, &sAh[row * LDT + kb + 2], &sAl[row * LDT + kb + 2]);
      split2(v.w, &sAh[row * LDT + kb + 3], &sAl[row * LDT + kb + 3]);
    }
    // stage B: W_upd tile 32x128, transposed into [n][k]
    for (int p = 0; p < 8; ++p) {
      const int idx = p * 128 + t;
      const int n4 = (idx & 31) * 4;
      const int k = idx >> 5;
      const float4 v = *(const float4*)&Wupd[(size_t)(k0 + k) * DF + n4];
      split2(v.x, &sBh[(n4 + 0) * LDT + k], &sBl[(n4 + 0) * LDT + k]);
      split2(v.y, &sBh[(n4 + 1) * LDT + k], &sBl[(n4 + 1) * LDT + k]);
      split2(v.z, &sBh[(n4 + 2) * LDT + k], &sBl[(n4 + 2) * LDT + k]);
      split2(v.w, &sBh[(n4 + 3) * LDT + k], &sBl[(n4 + 3) * LDT + k]);
    }
    __syncthreads();
    const v16bf ah = frag_A(&sAh[(wave * 16 + m) * LDT], h);
    const v16bf al = frag_A(&sAl[(wave * 16 + m) * LDT], h);
    #pragma unroll
    for (int ut = 0; ut < 8; ++ut) {
      const v16bf bh = frag_B(&sBh[(ut * 16 + m) * LDT], h);
      const v16bf bl = frag_B(&sBl[(ut * 16 + m) * LDT], h);
      acc[ut] = wmma_bf(ah, bh, acc[ut]);
      acc[ut] = wmma_bf(ah, bl, acc[ut]);
      acc[ut] = wmma_bf(al, bh, acc[ut]);
    }
  }
  #pragma unroll
  for (int ut = 0; ut < 8; ++ut) {
    const int u = ut * 16 + m;
    #pragma unroll
    for (int e = 0; e < 8; ++e) {
      const int rl = wave * 16 + e + 8 * h;
      out[((size_t)b * NN + r0 + rl) * DF + u] = acc[ut][e];
    }
  }
}

extern "C" void kernel_launch(void* const* d_in, const int* in_sizes, int n_in,
                              void* d_out, int out_size, void* d_ws, size_t ws_size,
                              hipStream_t stream) {
  const float* x    = (const float*)d_in[0];
  const float* adj  = (const float*)d_in[1];
  const float* Wmsg = (const float*)d_in[2];
  const float* Wupd = (const float*)d_in[3];
  float* out = (float*)d_out;

  char* ws = (char*)d_ws;
  const size_t nE = (size_t)NB * NN * DF;        // 2,097,152 elements
  bf16_t* xs_hi = (bf16_t*)(ws);                 // 2*nE bytes
  bf16_t* xs_lo = (bf16_t*)(ws + 2 * nE);        // 2*nE bytes
  float*  xr    = (float*) (ws + 4 * nE);        // 4*nE bytes
  float*  msg   = (float*) (ws + 8 * nE);        // 4*nE bytes  (total 24 MiB)

  dim3 grid(NB * (NN / 64));                     // 256 workgroups
  dim3 block(128);                               // 4 x wave32
  mpnn_k1_proj  <<<grid, block, 0, stream>>>(x, Wmsg, xs_hi, xs_lo, xr);
  mpnn_k2_gather<<<grid, block, 0, stream>>>(adj, xs_hi, xs_lo, xr, msg);
  mpnn_k3_update<<<grid, block, 0, stream>>>(x, msg, Wupd, out);
  (void)in_sizes; (void)n_in; (void)out_size; (void)ws_size;
}
